// TransformerLayerController_29076928593920
// MI455X (gfx1250) — matrix operand order repaired and checked by reading, not compiled, not run on hardware
//
#include <hip/hip_runtime.h>

// ---------------------------------------------------------------------------
// KIVI-style 4-bit KV fake-quant + causal flash attention for MI455X (gfx1250)
// B=1, H=32, S=2048, D=128.  wave32, WMMA f16 16x16x32.
// Both K and V tiles staged by the Tensor Data Mover (TENSORcnt), V via a
// 2-D strided descriptor from a channel-major V copy; double-buffered and
// software-pipelined so the DMA hides behind the WMMA work.
// ---------------------------------------------------------------------------

typedef __attribute__((ext_vector_type(16))) _Float16 v16h;
typedef __attribute__((ext_vector_type(8)))  float    v8f;
typedef __attribute__((ext_vector_type(4)))  unsigned int u32x4;
typedef __attribute__((ext_vector_type(8)))  int      i32x8;
typedef __attribute__((ext_vector_type(4)))  int      i32x4;
typedef __attribute__((address_space(3)))    _Float16 lds_f16_t;

#define H_    32
#define S_    2048
#define D_    128
#define QMAXF 7.0f
#define NOUT  16
#define BM    64
#define BN    64
#define TILE_ELEMS (BN * D_)   // 8192 f16 elements = 16 KB per tile

#if __has_builtin(__builtin_amdgcn_tensor_load_to_lds)
#define HAVE_TDM 1
#else
#define HAVE_TDM 0
#endif

union AB { v16h v; _Float16 h[16]; unsigned int u[8]; };
union C8 { v8f  v; float f[8]; };

#if HAVE_TDM
__device__ __forceinline__ unsigned lds_off(const _Float16* p) {
  return (unsigned)(unsigned long long)(const lds_f16_t*)p;
}
// Issue a TDM load: 2-D tile (t0 x t1 elements, 2 bytes each, row stride s0)
// from global into LDS (packed row-major).  Descriptor per ISA 8.3/8.4.
__device__ __forceinline__ void tdm_load(unsigned lds_addr, const void* gptr,
                                         unsigned t_dim0, unsigned t_dim1,
                                         unsigned tile0, unsigned tile1,
                                         unsigned stride0) {
  const unsigned long long ga = (unsigned long long)gptr;
  u32x4 g0;
  g0[0] = 1u;                                    // count=1 (valid D#)
  g0[1] = lds_addr;                              // lds_addr (bytes)
  g0[2] = (unsigned)ga;                          // global_addr[31:0]
  g0[3] = ((unsigned)(ga >> 32) & 0x1FFFFFFu)    // global_addr[56:32]
          | 0x80000000u;                         // type=2 ("image")
  i32x8 g1;
  g1[0] = 0x10000;                               // data_size=1 (2 bytes)
  g1[1] = (int)(t_dim0 << 16);                   // tensor_dim0[15:0]
  g1[2] = (int)((t_dim0 >> 16) | (t_dim1 << 16));// dim0[31:16] | dim1[15:0]
  g1[3] = (int)((t_dim1 >> 16) | (tile0 << 16)); // dim1[31:16] | tile_dim0
  g1[4] = (int)(tile1 & 0xFFFFu);                // tile_dim1 (tile_dim2=0)
  g1[5] = (int)stride0;                          // tensor_dim0_stride[31:0]
  g1[6] = 0;
  g1[7] = 0;
  i32x4 z4 = {0, 0, 0, 0};
#if defined(__clang_major__) && __clang_major__ >= 23
  i32x8 z8 = {0, 0, 0, 0, 0, 0, 0, 0};
  __builtin_amdgcn_tensor_load_to_lds(g0, g1, z4, z4, z8, 0);
#else
  __builtin_amdgcn_tensor_load_to_lds(g0, g1, z4, z4, 0);
#endif
}
#endif

// ---------------------------------------------------------------------------
// K quant: per (h,d) row along tokens (length S, stride D).
// 17 argmax passes: first 16 -> outliers (kept exact), 17th -> dense absmax.
// Output layout: [h][s][d] f16 (row-major per head, as attention consumes it).
// ---------------------------------------------------------------------------
__global__ __launch_bounds__(256) void kquant_kernel(const float* __restrict__ K,
                                                     _Float16* __restrict__ Kq) {
  __shared__ float absv[S_];
  __shared__ float red[256];
  __shared__ int   redi[256];
  __shared__ float s_thr;
  const int t = threadIdx.x;
  const int row = blockIdx.x;            // h*D + d
  const int h = row / D_, d = row % D_;
  const float* base = K + (size_t)h * S_ * D_ + d;

  for (int i = t; i < S_; i += 256) absv[i] = fabsf(base[(size_t)i * D_]);
  __syncthreads();

  for (int it = 0; it < NOUT + 1; ++it) {
    float m = -1.0f; int mi = 0;
    for (int i = t; i < S_; i += 256) { float a = absv[i]; if (a > m) { m = a; mi = i; } }
    red[t] = m; redi[t] = mi;
    __syncthreads();
    for (int off = 128; off > 0; off >>= 1) {
      if (t < off && red[t + off] > red[t]) { red[t] = red[t + off]; redi[t] = redi[t + off]; }
      __syncthreads();
    }
    if (t == 0) { if (it < NOUT) absv[redi[0]] = -1.0f; else s_thr = red[0]; }
    __syncthreads();
  }

  const float scale = fmaxf(s_thr, 1e-6f) * (1.0f / QMAXF);
  const float inv   = 1.0f / scale;
  for (int i = t; i < S_; i += 256) {
    float x = base[(size_t)i * D_];
    float y;
    if (absv[i] < 0.0f) {
      y = x;                                     // isolated outlier: exact
    } else {
      float q = rintf(x * inv);
      q = fminf(QMAXF, fmaxf(-QMAXF, q));
      y = q * scale;
    }
    Kq[(size_t)h * S_ * D_ + (size_t)i * D_ + d] = (_Float16)y;
  }
}

// ---------------------------------------------------------------------------
// V quant: per (h,s) row along channels (length D=128). One wave per row.
// Output layout: [h][channel][token] f16 (channel-major) so the attention
// kernel's V tile is a plain 2-D strided TDM descriptor (no transpose loop).
// ---------------------------------------------------------------------------
__global__ __launch_bounds__(256) void vquant_kernel(const float* __restrict__ V,
                                                     _Float16* __restrict__ Vq) {
  const int lane = threadIdx.x & 31;
  const int wv   = threadIdx.x >> 5;
  const long row = (long)blockIdx.x * 8 + wv;     // < H*S
  const int  h   = (int)(row / S_);
  const int  ss  = (int)(row % S_);
  const float* base = V + row * D_;

  float x[4], a[4];
#pragma unroll
  for (int j = 0; j < 4; ++j) { x[j] = base[lane + 32 * j]; a[j] = fabsf(x[j]); }

  float thr = 0.0f;
  for (int it = 0; it < NOUT + 1; ++it) {
    float m = a[0]; int mj = 0;
#pragma unroll
    for (int j = 1; j < 4; ++j) if (a[j] > m) { m = a[j]; mj = j; }
    int midx = lane + 32 * mj;
#pragma unroll
    for (int off = 16; off > 0; off >>= 1) {
      float om = __shfl_xor(m, off, 32);
      int   oi = __shfl_xor(midx, off, 32);
      if (om > m) { m = om; midx = oi; }
    }
    if (it < NOUT) { if ((midx & 31) == lane) a[midx >> 5] = -1.0f; }
    else thr = m;
  }

  const float scale = fmaxf(thr, 1e-6f) * (1.0f / QMAXF);
  const float inv   = 1.0f / scale;
#pragma unroll
  for (int j = 0; j < 4; ++j) {
    float y;
    if (a[j] < 0.0f) {
      y = x[j];
    } else {
      float q = rintf(x[j] * inv);
      q = fminf(QMAXF, fmaxf(-QMAXF, q));
      y = q * scale;
    }
    const int c = lane + 32 * j;
    Vq[((size_t)h * D_ + c) * S_ + ss] = (_Float16)y;   // transposed store
  }
}

// ---------------------------------------------------------------------------
// Flash attention over reconstructed f16 KV.
// Block = 128 threads (4 wave32). BLOCK_M=BLOCK_N=64. Each wave: 16 Q rows.
// Double-buffered TDM staging pipelined against the WMMA compute.
// ---------------------------------------------------------------------------
__global__ __launch_bounds__(128) void attn_kernel(const float* __restrict__ Q,
                                                   const _Float16* __restrict__ Kq,
                                                   const _Float16* __restrict__ Vq,
                                                   float* __restrict__ O) {
  __shared__ _Float16 qs[BM * D_];         // Q tile (dead after qa hoist)
  __shared__ _Float16 kt[2][BN * D_];      // K tiles [token][channel]
  __shared__ _Float16 vt[2][D_ * BN];      // V tiles [channel][token]
  __shared__ _Float16 ps[4 * 16 * BN];     // per-wave P tiles (C->A relayout)

  const int tid  = threadIdx.x;
  const int lane = tid & 31;
  const int w    = tid >> 5;            // wave 0..3
  const int nn   = lane & 15;           // N / M-fragment lane index
  const int hi   = lane >> 4;           // lane half

  const int h     = blockIdx.y;
  const int qTile = blockIdx.x;
  const int qbase = qTile * BM;
  const int nTiles = qTile + 1;         // causal: only tiles at/below diagonal

  const float*    Qh = Q  + (size_t)h * S_ * D_;
  const _Float16* Kh = Kq + (size_t)h * S_ * D_;
  const _Float16* Vh = Vq + (size_t)h * D_ * S_;   // channel-major per head

#if HAVE_TDM
  unsigned ktl[2], vtl[2];
  ktl[0] = lds_off(kt[0]); ktl[1] = lds_off(kt[1]);
  vtl[0] = lds_off(vt[0]); vtl[1] = lds_off(vt[1]);

  // Kick off tile 0 DMAs immediately (overlaps the Q staging below).
  if (tid < 32) {
    tdm_load(ktl[0], Kh, TILE_ELEMS, 1, TILE_ELEMS, 0, TILE_ELEMS);     // 1-D
    tdm_load(vtl[0], Vh, S_, D_, BN, D_, S_);                           // 2-D
  }
#else
  {
    const unsigned int* src = (const unsigned int*)Kh;
    unsigned int*       dst = (unsigned int*)kt[0];
    for (int i = tid; i < (BN * D_) / 2; i += 128) dst[i] = src[i];
    for (int i = tid; i < BN * D_; i += 128)
      vt[0][i] = Vh[(size_t)(i >> 6) * S_ + (i & 63)];
  }
#endif

  // Stage Q tile (fold in 1/sqrt(D))
  for (int i = tid; i < BM * D_; i += 128)
    qs[i] = (_Float16)(Qh[(size_t)qbase * D_ + i] * 0.08838834764831845f);
  __syncthreads();

  // Q A-fragments are invariant across the KV loop: hoist into registers.
  AB qa[4];
#pragma unroll
  for (int ks = 0; ks < 4; ++ks) {
    const unsigned int* ap = (const unsigned int*)(qs + (w * 16 + nn) * D_ + ks * 32);
#pragma unroll
    for (int v = 0; v < 8; ++v) qa[ks].u[v] = ap[(v & 3) + hi * 4 + ((v >> 2) << 3)];
  }

#if HAVE_TDM
  if (tid < 32) __builtin_amdgcn_s_wait_tensorcnt(0);   // tile 0 landed
#endif
  __syncthreads();

  C8 o[8];
#pragma unroll
  for (int dt = 0; dt < 8; ++dt)
#pragma unroll
    for (int r = 0; r < 8; ++r) o[dt].f[r] = 0.0f;

  float rowm[8], lsum[8];
#pragma unroll
  for (int r = 0; r < 8; ++r) { rowm[r] = -1e30f; lsum[r] = 0.0f; }

  for (int kv = 0; kv < nTiles; ++kv) {
    const int kvbase = kv * BN;
    const int cur = kv & 1, nxt = cur ^ 1;

    // ---- issue next tile's DMAs into the other buffer (fully overlapped) --
    if (kv + 1 < nTiles) {
#if HAVE_TDM
      if (tid < 32) {
        tdm_load(ktl[nxt], Kh + (size_t)(kvbase + BN) * D_,
                 TILE_ELEMS, 1, TILE_ELEMS, 0, TILE_ELEMS);
        tdm_load(vtl[nxt], Vh + (kvbase + BN),
                 S_, D_, BN, D_, S_);
      }
      if (kv + 2 < nTiles) {   // warm GL2 ahead of the following DMA
        __builtin_prefetch(Kh + (size_t)(kvbase + 2 * BN) * D_ + tid * 64, 0, 1);
      }
#else
      const unsigned int* src = (const unsigned int*)(Kh + (size_t)(kvbase + BN) * D_);
      unsigned int*       dst = (unsigned int*)kt[nxt];
      for (int i = tid; i < (BN * D_) / 2; i += 128) dst[i] = src[i];
      for (int i = tid; i < BN * D_; i += 128)
        vt[nxt][i] = Vh[(size_t)(i >> 6) * S_ + kvbase + BN + (i & 63)];
#endif
    }

    // ---- S = Q * K^T : 4 n-subtiles x 4 k-steps = 16 WMMA per wave ----
    C8 s[4];
#pragma unroll
    for (int t = 0; t < 4; ++t)
#pragma unroll
      for (int r = 0; r < 8; ++r) s[t].f[r] = 0.0f;

#pragma unroll
    for (int t = 0; t < 4; ++t) {
      const unsigned int* bp = (const unsigned int*)(kt[cur] + (t * 16 + nn) * D_);
#pragma unroll
      for (int ks = 0; ks < 4; ++ks) {
        AB b;
#pragma unroll
        for (int v = 0; v < 8; ++v)           // B f16 32x16 layout
          b.u[v] = bp[ks * 16 + (v & 3) + hi * 4 + ((v >> 2) << 3)];
        s[t].v = __builtin_amdgcn_wmma_f32_16x16x32_f16(
            false, qa[ks].v, false, b.v, (short)0, s[t].v, false, false);
      }
    }

    // ---- causal mask (diagonal tile only) ----
    if (kvbase + BN > qbase) {
#pragma unroll
      for (int t = 0; t < 4; ++t) {
        int col = kvbase + t * 16 + nn;
#pragma unroll
        for (int r = 0; r < 8; ++r) {
          int qr = qbase + w * 16 + r + hi * 8;
          if (col > qr) s[t].f[r] = -1e30f;
        }
      }
    }

    // ---- online softmax (per-row stats across 16-lane groups) ----
    float corr[8];
#pragma unroll
    for (int r = 0; r < 8; ++r) {
      float v = fmaxf(fmaxf(s[0].f[r], s[1].f[r]), fmaxf(s[2].f[r], s[3].f[r]));
#pragma unroll
      for (int off = 1; off < 16; off <<= 1) v = fmaxf(v, __shfl_xor(v, off, 16));
      float mnew = fmaxf(rowm[r], v);
      corr[r] = __expf(rowm[r] - mnew);
      rowm[r] = mnew;
    }
#pragma unroll
    for (int t = 0; t < 4; ++t)
#pragma unroll
      for (int r = 0; r < 8; ++r) s[t].f[r] = __expf(s[t].f[r] - rowm[r]);
#pragma unroll
    for (int r = 0; r < 8; ++r) {
      float sum = s[0].f[r] + s[1].f[r] + s[2].f[r] + s[3].f[r];
#pragma unroll
      for (int off = 1; off < 16; off <<= 1) sum += __shfl_xor(sum, off, 16);
      lsum[r] = lsum[r] * corr[r] + sum;
    }
#pragma unroll
    for (int dt = 0; dt < 8; ++dt)
#pragma unroll
      for (int r = 0; r < 8; ++r) o[dt].f[r] *= corr[r];

    // ---- P: C-layout -> LDS -> A-layout (wave-private region) ----
    _Float16* pw = ps + w * 16 * BN;
#pragma unroll
    for (int t = 0; t < 4; ++t)
#pragma unroll
      for (int r = 0; r < 8; ++r)
        pw[(r + hi * 8) * BN + t * 16 + nn] = (_Float16)s[t].f[r];

    // ---- O += P * V : 2 k-slices x 8 d-subtiles = 16 WMMA per wave ----
#pragma unroll
    for (int js = 0; js < 2; ++js) {
      AB a;
      const unsigned int* ap = (const unsigned int*)(pw + nn * BN + js * 32);
#pragma unroll
      for (int v = 0; v < 8; ++v)
        a.u[v] = ap[(v & 3) + hi * 4 + ((v >> 2) << 3)];
#pragma unroll
      for (int dt = 0; dt < 8; ++dt) {
        AB b;
        const unsigned int* bp =
            (const unsigned int*)(vt[cur] + (dt * 16 + nn) * BN + js * 32);
#pragma unroll
        for (int v = 0; v < 8; ++v) b.u[v] = bp[(v & 3) + hi * 4 + ((v >> 2) << 3)];
        o[dt].v = __builtin_amdgcn_wmma_f32_16x16x32_f16(
            false, a.v, false, b.v, (short)0, o[dt].v, false, false);
      }
    }

    // ---- close the pipeline stage: next-tile DMAs done, publish LDS ----
#if HAVE_TDM
    if (tid < 32) __builtin_amdgcn_s_wait_tensorcnt(0);
#endif
    __syncthreads();
  }

  // ---- epilogue: normalize and write f32 output [B,H,S,D] ----
  float rcp[8];
#pragma unroll
  for (int r = 0; r < 8; ++r) rcp[r] = 1.0f / lsum[r];
  float* Oh = O + (size_t)h * S_ * D_;
#pragma unroll
  for (int dt = 0; dt < 8; ++dt)
#pragma unroll
    for (int r = 0; r < 8; ++r) {
      int rowg = qbase + w * 16 + r + hi * 8;
      Oh[(size_t)rowg * D_ + dt * 16 + nn] = o[dt].f[r] * rcp[r];
    }
}

// ---------------------------------------------------------------------------
extern "C" void kernel_launch(void* const* d_in, const int* in_sizes, int n_in,
                              void* d_out, int out_size, void* d_ws, size_t ws_size,
                              hipStream_t stream) {
  (void)in_sizes; (void)n_in; (void)out_size; (void)ws_size;
  const float* q = (const float*)d_in[0];
  const float* k = (const float*)d_in[1];
  const float* v = (const float*)d_in[2];
  float* out = (float*)d_out;

  _Float16* Kq = (_Float16*)d_ws;                       // 16 MB  [h][s][d]
  _Float16* Vq = Kq + (size_t)H_ * S_ * D_;             // +16 MB [h][d][s]

  hipLaunchKernelGGL(kquant_kernel, dim3(H_ * D_), dim3(256), 0, stream, k, Kq);
  hipLaunchKernelGGL(vquant_kernel, dim3((H_ * S_) / 8), dim3(256), 0, stream, v, Vq);
  hipLaunchKernelGGL(attn_kernel, dim3(S_ / BM, H_), dim3(128), 0, stream, q, Kq, Vq, out);
}
